// NaiveMHLA_14147622273638
// MI455X (gfx1250) — matrix operand order repaired and checked by reading, not compiled, not run on hardware
//
#include <hip/hip_runtime.h>

// ---------------------------------------------------------------------------
// MLA forward for MI455X (gfx1250): bf16 WMMA (f32 accum) everywhere that
// matters, flash-attention fusion so (B,NH,T,T) scores never hit memory.
// KV tiles are staged into LDS with the gfx1250 async-to-LDS engine
// (GLOBAL_LOAD_ASYNC_TO_LDS_B128 + s_wait_asynccnt), no VGPR round-trip.
// ---------------------------------------------------------------------------

#define BATCH 2
#define T_DIM 2048
#define C_DIM 1024
#define NHD   16
#define HSZ   64
#define KVLAT 128

typedef unsigned short u16;
typedef __attribute__((ext_vector_type(4)))  unsigned int u32x4;
typedef __attribute__((ext_vector_type(16))) __bf16       bf16x16;
typedef __attribute__((ext_vector_type(8)))  float        f32x8;

union Frag {            // one WMMA 16-bit operand fragment: 16 bf16 per lane
    u32x4   q[2];       // two 16-byte chunks (two *_b128 loads)
    bf16x16 v;
};

__device__ __forceinline__ u16 f2bf(float f) {
    unsigned u = __builtin_bit_cast(unsigned, f);
    unsigned r = u + 0x7FFFu + ((u >> 16) & 1u);   // round-to-nearest-even
    return (u16)(r >> 16);
}

// ---------------------------------------------------------------------------
// fp32 -> bf16 conversion
// ---------------------------------------------------------------------------
__global__ __launch_bounds__(256) void cvt_bf16_k(const float* __restrict__ in,
                                                  u16* __restrict__ out, int n) {
    int i = blockIdx.x * 256 + threadIdx.x;
    if (i < n) out[i] = f2bf(in[i]);
}

// ---------------------------------------------------------------------------
// Tiny fp32 GEMM for the absorbed weight matrices (<0.5 GFLOP total).
// C[m][n] = sum_k A[m*asm + k*ask] * B[k*bsk + n*bsn]
// ---------------------------------------------------------------------------
__global__ __launch_bounds__(256) void gemm_small_k(
        const float* __restrict__ A, int asm_, int ask,
        const float* __restrict__ B, int bsk, int bsn,
        float* outF, u16* outB, int N, int K, int total) {
    int idx = blockIdx.x * 256 + threadIdx.x;
    if (idx >= total) return;
    int m = idx / N, n = idx - m * N;
    float acc = 0.f;
    for (int k = 0; k < K; ++k)
        acc = fmaf(A[(size_t)m * asm_ + (size_t)k * ask],
                   B[(size_t)k * bsk + (size_t)n * bsn], acc);
    if (outF) outF[idx] = acc;
    if (outB) outB[idx] = f2bf(acc);
}

// ---------------------------------------------------------------------------
// Generic WMMA GEMM:  C[M x N] = A[M x K] * Bt[N x K]^T   (all bf16 operands)
// 128 threads = 4 waves; 64x64 tile per block; 16x16x32 bf16 WMMA.
// A-fragment: lane m = lane&15, chunks at k0+hi*8 and k0+16+hi*8 (ISA layout).
// B-fragment: lane n = lane&15, chunks at k0+hi*8 and k0+16+hi*8 (ISA layout).
// grid.z batches with element strides aZ/bZ/oZ (used for per-head ql GEMM).
// ---------------------------------------------------------------------------
__global__ __launch_bounds__(128) void gemm_bt_k(
        const u16* __restrict__ A, int lda, int aZ,
        const u16* __restrict__ Bt, int ldb, int bZ,
        float* outF, u16* outB, int ldo, int oZ,
        int K) {
    A  += (size_t)blockIdx.z * aZ;
    Bt += (size_t)blockIdx.z * bZ;
    const int wave = threadIdx.x >> 5;
    const int lane = threadIdx.x & 31;
    const int hi   = lane >> 4;
    const int ln   = lane & 15;
    const int m0 = blockIdx.y * 64 + wave * 16;
    const int n0 = blockIdx.x * 64;

    f32x8 acc[4] = {};
    for (int k0 = 0; k0 < K; k0 += 32) {
        Frag a;
        const u16* ar = A + (size_t)(m0 + ln) * lda + k0;
        a.q[0] = *(const u32x4*)(ar + hi * 8);
        a.q[1] = *(const u32x4*)(ar + 16 + hi * 8);
#pragma unroll
        for (int j = 0; j < 4; ++j) {
            Frag b;
            const u16* br = Bt + (size_t)(n0 + j * 16 + ln) * ldb + k0;
            b.q[0] = *(const u32x4*)(br + hi * 8);
            b.q[1] = *(const u32x4*)(br + 16 + hi * 8);
            acc[j] = __builtin_amdgcn_wmma_f32_16x16x32_bf16(
                false, a.v, false, b.v, (short)0, acc[j], false, false);
        }
    }
#pragma unroll
    for (int j = 0; j < 4; ++j)
#pragma unroll
        for (int r = 0; r < 8; ++r) {
            size_t o = (size_t)(m0 + hi * 8 + r) * ldo + n0 + j * 16 + ln
                     + (size_t)blockIdx.z * oZ;
            float v = acc[j][r];
            if (outF) outF[o] = v;
            if (outB) outB[o] = f2bf(v);
        }
}

// ---------------------------------------------------------------------------
// Fused flash-attention + output projection.
// grid = (T/64, NH, B); block = 128 threads (4 waves, 16 query rows each).
//   S   = ql_tile(64x128) @ c_kv_tile(64x128)^T   (bf16 WMMA, fp32 acc)
//   online causal softmax (16-lane shfl_xor row reductions)
//   ctx += P(16x64) @ c_kv_tile(64x128)           (via transposed LDS copy)
//   y    = (ctx / l) (16x128) @ v_effT_h(64x128)^T
// KV tile comes in through the async-to-LDS pipe (ASYNCcnt), then the
// transposed copy is built LDS->LDS through the DS pipe.
// LDS strides padded so every ds_load_b128 is 16B-aligned and bank-spread.
// ---------------------------------------------------------------------------
#define KVS_STR 136   // 64 x 128 KV tile, row-major (s, l)
#define KVT_STR 72    // 128 x 64 transposed tile (l, s)
#define PS_STR  136   // per-wave 16 x 128 staging (P, then ctx)

__global__ __launch_bounds__(128) void mla_attn_k(
        const u16* __restrict__ qlb,   // (B, T, NH, KVL) bf16
        const u16* __restrict__ ckvb,  // (B, T, KVL)     bf16
        const u16* __restrict__ vt,    // (C, KVL) bf16 : v_effT = W_o @ W_uv
        float* __restrict__ y) {       // (B, T, C) fp32
    __shared__ u16 kvS[64 * KVS_STR];
    __shared__ u16 kvT[128 * KVT_STR];
    __shared__ u16 pS[4][16 * PS_STR];

    const int t0   = blockIdx.x * 64;
    const int h    = blockIdx.y;
    const int bb   = blockIdx.z;
    const int tid  = threadIdx.x;
    const int wave = tid >> 5;
    const int lane = tid & 31;
    const int hi   = lane >> 4;
    const int ln   = lane & 15;

    // Q fragments for this wave's 16 rows (K = 128 -> 4 k-steps), kept live.
    Frag qa[4];
    {
        const u16* qr = qlb + (((size_t)(bb * T_DIM + t0 + wave * 16 + ln)) * NHD + h) * KVLAT;
#pragma unroll
        for (int ks = 0; ks < 4; ++ks) {
            qa[ks].q[0] = *(const u32x4*)(qr + ks * 32 + hi * 8);
            qa[ks].q[1] = *(const u32x4*)(qr + ks * 32 + 16 + hi * 8);
        }
    }

    f32x8 ctx[8] = {};
    float mrow[8], lrow[8];
#pragma unroll
    for (int r = 0; r < 8; ++r) { mrow[r] = -1e30f; lrow[r] = 0.f; }

    const u16* ckv_b = ckvb + (size_t)bb * T_DIM * KVLAT;
    const int nTiles = t0 / 64 + 1;   // causal: only tiles with s0 <= t0

    for (int tile = 0; tile < nTiles; ++tile) {
        const int s0 = tile * 64;
        __syncthreads();              // previous tile's LDS reads done

        // ---- async copy of the 64x128 bf16 KV tile straight into LDS ----
        // 1024 chunks of 16B; each thread issues 8 GLOBAL_LOAD_ASYNC_TO_LDS_B128.
        // Per ISA 10.2, low 32 bits of a generic LDS pointer are the LDS byte
        // offset, which is what the VDST operand of the async load wants.
#pragma unroll
        for (int rep = 0; rep < 8; ++rep) {
            int chunk = rep * 128 + tid;
            int srow  = chunk >> 4;
            int cb    = (chunk & 15) * 8;
            unsigned long long ga =
                (unsigned long long)(ckv_b + (size_t)(s0 + srow) * KVLAT + cb);
            unsigned lds_off = (unsigned)(size_t)&kvS[srow * KVS_STR + cb];
            asm volatile("global_load_async_to_lds_b128 %0, %1, off"
                         :: "v"(lds_off), "v"(ga) : "memory");
        }
        asm volatile("s_wait_asynccnt 0x0" ::: "memory");
        __syncthreads();              // tile visible to all waves

        // ---- build transposed copy (l, s) LDS->LDS through the DS pipe ----
#pragma unroll
        for (int rep = 0; rep < 8; ++rep) {
            int chunk = rep * 128 + tid;
            int srow  = chunk >> 4;
            int cb    = (chunk & 15) * 8;
            u32x4 v = *(const u32x4*)&kvS[srow * KVS_STR + cb];
            const u16* e = (const u16*)&v;
#pragma unroll
            for (int q = 0; q < 8; ++q) kvT[(cb + q) * KVT_STR + srow] = e[q];
        }
        __syncthreads();

        if (tile + 1 < nTiles)        // gfx1250 global_prefetch_b8, next tile
            __builtin_prefetch(ckv_b + (size_t)(s0 + 64) * KVLAT + tid * 64, 0, 0);

        // S = Q @ K^T : per-wave 16x64, K-dim 128
        f32x8 sacc[4] = {};
#pragma unroll
        for (int ks = 0; ks < 4; ++ks)
#pragma unroll
            for (int j = 0; j < 4; ++j) {
                Frag b;
                const u16* br = &kvS[(j * 16 + ln) * KVS_STR + ks * 32];
                b.q[0] = *(const u32x4*)(br + hi * 8);
                b.q[1] = *(const u32x4*)(br + 16 + hi * 8);
                sacc[j] = __builtin_amdgcn_wmma_f32_16x16x32_bf16(
                    false, qa[ks].v, false, b.v, (short)0, sacc[j], false, false);
            }

        // online softmax (rows spread over 16 lanes; acc row = hi*8 + r)
        const bool diag = (s0 == t0);
#pragma unroll
        for (int r = 0; r < 8; ++r) {
            int wrow = wave * 16 + hi * 8 + r;
            float mx = -1e30f;
#pragma unroll
            for (int j = 0; j < 4; ++j) {
                float v = sacc[j][r] * 0.125f;               // 1/sqrt(HS)
                if (diag && (j * 16 + ln > wrow)) v = -1e30f; // causal mask
                sacc[j][r] = v;
                mx = fmaxf(mx, v);
            }
            mx = fmaxf(mx, __shfl_xor(mx, 1));
            mx = fmaxf(mx, __shfl_xor(mx, 2));
            mx = fmaxf(mx, __shfl_xor(mx, 4));
            mx = fmaxf(mx, __shfl_xor(mx, 8));
            float mn    = fmaxf(mrow[r], mx);
            float alpha = __expf(mrow[r] - mn);
            mrow[r] = mn;
            float ls = 0.f;
#pragma unroll
            for (int j = 0; j < 4; ++j) {
                float p = __expf(sacc[j][r] - mn);
                pS[wave][(hi * 8 + r) * PS_STR + j * 16 + ln] = f2bf(p);
                ls += p;
            }
            ls += __shfl_xor(ls, 1);
            ls += __shfl_xor(ls, 2);
            ls += __shfl_xor(ls, 4);
            ls += __shfl_xor(ls, 8);
            lrow[r] = lrow[r] * alpha + ls;
#pragma unroll
            for (int jj = 0; jj < 8; ++jj) ctx[jj][r] *= alpha;
        }

        // ctx += P(16x64) @ KV(64x128)  (B fragments from transposed tile)
#pragma unroll
        for (int ks = 0; ks < 2; ++ks) {
            Frag a;
            const u16* ar = &pS[wave][ln * PS_STR + ks * 32];
            a.q[0] = *(const u32x4*)(ar + hi * 8);
            a.q[1] = *(const u32x4*)(ar + 16 + hi * 8);
#pragma unroll
            for (int jj = 0; jj < 8; ++jj) {
                Frag b;
                const u16* br = &kvT[(jj * 16 + ln) * KVT_STR + ks * 32];
                b.q[0] = *(const u32x4*)(br + hi * 8);
                b.q[1] = *(const u32x4*)(br + 16 + hi * 8);
                ctx[jj] = __builtin_amdgcn_wmma_f32_16x16x32_bf16(
                    false, a.v, false, b.v, (short)0, ctx[jj], false, false);
            }
        }
    }

    // normalize and stage ctx (16x128) as bf16 A-operand in per-wave LDS
#pragma unroll
    for (int r = 0; r < 8; ++r) {
        float inv = 1.0f / lrow[r];
#pragma unroll
        for (int jj = 0; jj < 8; ++jj)
            pS[wave][(hi * 8 + r) * PS_STR + jj * 16 + ln] = f2bf(ctx[jj][r] * inv);
    }

    // y_tile(16x64) = ctx(16x128) @ v_effT_h(64x128)^T ; write fp32 y
#pragma unroll
    for (int jn = 0; jn < 4; ++jn) {
        f32x8 yacc = {};
#pragma unroll
        for (int ks = 0; ks < 4; ++ks) {
            Frag a;
            const u16* ar = &pS[wave][ln * PS_STR + ks * 32];
            a.q[0] = *(const u32x4*)(ar + hi * 8);
            a.q[1] = *(const u32x4*)(ar + 16 + hi * 8);
            Frag b;
            const u16* br = vt + (size_t)(h * HSZ + jn * 16 + ln) * KVLAT + ks * 32;
            b.q[0] = *(const u32x4*)(br + hi * 8);
            b.q[1] = *(const u32x4*)(br + 16 + hi * 8);
            yacc = __builtin_amdgcn_wmma_f32_16x16x32_bf16(
                false, a.v, false, b.v, (short)0, yacc, false, false);
        }
#pragma unroll
        for (int r = 0; r < 8; ++r)
            y[((size_t)bb * T_DIM + t0 + wave * 16 + hi * 8 + r) * C_DIM
              + h * HSZ + jn * 16 + ln] = yacc[r];
    }
}

// ---------------------------------------------------------------------------
extern "C" void kernel_launch(void* const* d_in, const int* in_sizes, int n_in,
                              void* d_out, int out_size, void* d_ws, size_t ws_size,
                              hipStream_t stream) {
    const float* x     = (const float*)d_in[0];   // (B, T, C)
    const float* W_dq  = (const float*)d_in[1];   // (QL, C)  = (128, 1024)
    const float* W_uq  = (const float*)d_in[2];   // (C, QL)  = (1024, 128)
    const float* W_dkv = (const float*)d_in[3];   // (KVL, C) = (128, 1024)
    const float* W_uk  = (const float*)d_in[4];   // (C, KVL) = (1024, 128)
    const float* W_uv  = (const float*)d_in[5];   // (C, KVL) = (1024, 128)
    const float* W_o   = (const float*)d_in[6];   // (C, C)

    float* y_out   = (float*)d_out;                                  // B*T*C
    float* ckv_out = (float*)d_out + (size_t)BATCH * T_DIM * C_DIM;  // B*T*KVL

    // workspace carve-up (256B aligned)
    char* base = (char*)d_ws;
    size_t off = 0;
    auto carve = [&](size_t bytes) {
        void* p = base + off;
        off = (off + bytes + 255) & ~(size_t)255;
        return p;
    };
    const size_t BT = (size_t)BATCH * T_DIM;                 // 4096 rows
    u16*   xb    = (u16*)carve(BT * C_DIM * 2);              // x bf16
    u16*   wdqb  = (u16*)carve(128 * 1024 * 2);              // W_dq bf16
    u16*   wuqb  = (u16*)carve(1024 * 128 * 2);              // W_uq bf16
    u16*   wdkvb = (u16*)carve(128 * 1024 * 2);              // W_dkv bf16
    float* Gf    = (float*)carve(128 * 128 * 4);             // W_uk^T @ W_uq
    u16*   ket   = (u16*)carve(128 * 1024 * 2);              // k_eff^T (KVL, C)
    u16*   vtb   = (u16*)carve(1024 * 128 * 2);              // v_effT = W_o@W_uv
    u16*   tmpb  = (u16*)carve(BT * 128 * 2);                // x @ W_dq^T
    u16*   qfb   = (u16*)carve(BT * C_DIM * 2);              // q_full bf16
    u16*   ckvb  = (u16*)carve(BT * KVLAT * 2);              // c_kv bf16
    u16*   qlb   = (u16*)carve(BT * NHD * KVLAT * 2);        // ql (B,T,NH,KVL)

    // 1) fp32 -> bf16 conversions
    cvt_bf16_k<<<(int)(BT * C_DIM + 255) / 256, 256, 0, stream>>>(x, xb, (int)(BT * C_DIM));
    cvt_bf16_k<<<(128 * 1024 + 255) / 256, 256, 0, stream>>>(W_dq, wdqb, 128 * 1024);
    cvt_bf16_k<<<(1024 * 128 + 255) / 256, 256, 0, stream>>>(W_uq, wuqb, 1024 * 128);
    cvt_bf16_k<<<(128 * 1024 + 255) / 256, 256, 0, stream>>>(W_dkv, wdkvb, 128 * 1024);

    // 2) absorbed matrices (tiny):
    //    G[i][j]    = sum_c W_uk[c][i] * W_uq[c][j]            (128 x 128)
    gemm_small_k<<<64, 256, 0, stream>>>(W_uk, 1, 128, W_uq, 128, 1,
                                         Gf, nullptr, 128, 1024, 128 * 128);
    //    k_effT[l][c'] = sum_j G[l][j] * W_dq[j][c']           (128 x 1024)
    gemm_small_k<<<512, 256, 0, stream>>>(Gf, 128, 1, W_dq, 1024, 1,
                                          nullptr, ket, 1024, 128, 128 * 1024);
    //    v_effT[c'][l] = sum_c W_o[c'][c] * W_uv[c][l]         (1024 x 128)
    gemm_small_k<<<512, 256, 0, stream>>>(W_o, 1024, 1, W_uv, 128, 1,
                                          nullptr, vtb, 128, 1024, 1024 * 128);

    // 3) big projections on the WMMA pipe
    //    tmp = x @ W_dq^T                (4096 x 128, K=1024)
    gemm_bt_k<<<dim3(2, 64, 1), 128, 0, stream>>>(xb, 1024, 0, wdqb, 1024, 0,
                                                  nullptr, tmpb, 128, 0, 1024);
    //    c_kv = x @ W_dkv^T  -> fp32 to d_out tail + bf16 to ws
    gemm_bt_k<<<dim3(2, 64, 1), 128, 0, stream>>>(xb, 1024, 0, wdkvb, 1024, 0,
                                                  ckv_out, ckvb, 128, 0, 1024);
    //    q_full = tmp @ W_uq^T           (4096 x 1024, K=128)
    gemm_bt_k<<<dim3(16, 64, 1), 128, 0, stream>>>(tmpb, 128, 0, wuqb, 128, 0,
                                                   nullptr, qfb, 1024, 0, 128);
    //    ql[b,t,h,:] = q_full[:, h*64:+64] @ k_effT[:, h*64:+64]^T  per head
    gemm_bt_k<<<dim3(2, 64, NHD), 128, 0, stream>>>(qfb, 1024, HSZ, ket, 1024, HSZ,
                                                    nullptr, qlb, NHD * KVLAT, KVLAT, HSZ);

    // 4) fused flash attention + output projection
    mla_attn_k<<<dim3(T_DIM / 64, NHD, BATCH), 128, 0, stream>>>(qlb, ckvb, vtb, y_out);

    (void)in_sizes; (void)n_in; (void)out_size; (void)ws_size;
}
